// ActorCritic_81149112091129
// MI455X (gfx1250) — matrix operand order repaired and checked by reading, not compiled, not run on hardware
//
#include <hip/hip_runtime.h>
#include <stdint.h>

typedef _Float16 v16h __attribute__((ext_vector_type(16)));
typedef _Float16 v8h  __attribute__((ext_vector_type(8)));
typedef float    v8f  __attribute__((ext_vector_type(8)));

#define N_POINTS        262144
#define WAVES_PER_BLOCK 8
#define TILES_PER_WAVE  4      // 16-point tiles per wave
#define MAIN_BLOCKS     512    // 512 * 8 * 4 * 16 = 262144 points

union H2U    { uint32_t u; _Float16 h[2]; };
union F16x16 { v16h v; v8h h8[2]; uint32_t u[8]; _Float16 h[16]; };

// ---------------------------------------------------------------------------
// Kernel 0: build f16 weight images PRE-SWIZZLED into the WMMA A-fragment
// lane layout used here (contiguous k, 16x32 f16 A):
//   lane l: row M = l&15
//           halfs[0..7]  = K kc*32 + (l>=16?16:0) + 0..7
//           halfs[8..15] = K kc*32 + (l>=16?16:0) + 8..15
// Fragment image: sw[frag*512 + lane*16 + j]  (1 KB per fragment)
// sw2 (16 frags) and sw3 (64 frags) are CONTIGUOUS in d_ws so the main
// kernel can stream all 80 KB with one async-to-LDS loop.
// ---------------------------------------------------------------------------
__global__ void __launch_bounds__(256) prep_kernel(
    const float* __restrict__ c2w, const float* __restrict__ c3w,
    _Float16* __restrict__ sw2, _Float16* __restrict__ sw3,
    float* __restrict__ gpool)
{
  int idx = blockIdx.x * 256 + threadIdx.x;
  if (idx < 256) gpool[idx] = 0.0f;            // relu outputs >= 0
  if (idx < 8192) {                            // 16 frags (t2*2+kc) x 512
    int fid = idx >> 9, rem = idx & 511;
    int l = rem >> 4, j = rem & 15;
    int t2 = fid >> 1, kc = fid & 1;
    int row = t2 * 16 + (l & 15);
    int K = kc * 32 + ((l >> 4) << 4) + j;
    sw2[idx] = (_Float16)c2w[row * 64 + K];
  }
  if (idx < 32768) {                           // 64 frags (t3*4+kc) x 512
    int fid = idx >> 9, rem = idx & 511;
    int l = rem >> 4, j = rem & 15;
    int t3 = fid >> 2, kc = fid & 3;
    int row = t3 * 16 + (l & 15);
    int K = kc * 32 + ((l >> 4) << 4) + j;
    sw3[idx] = (_Float16)c3w[row * 128 + K];
  }
}

// ---------------------------------------------------------------------------
// Kernel 1: per-point MLP 2->64->128->256 (WMMA f16) + global max pool.
//   B-matrix 32x16 f16: lane l: col N = l&15; halfs[j] = K 16*(l>=16)+j
//   C/D 16x16 f32:      lane l: col N = l&15; VGPR v: row M = v + 8*(l>=16)
// Weight fragments staged to LDS with GLOBAL_LOAD_ASYNC_TO_LDS_B128
// (ASYNCcnt), then served as ds_load_b128.
// ---------------------------------------------------------------------------
__global__ void __launch_bounds__(256) pointnet_kernel(
    const float* __restrict__ x,
    const float* __restrict__ c1w, const float* __restrict__ c1b,
    const float* __restrict__ c2b, const float* __restrict__ c3b,
    const _Float16* __restrict__ swz,       // sw2 ++ sw3, 40960 halfs
    float* __restrict__ gpool)
{
  __shared__ __align__(16) _Float16 s_wh[40960];  // 80 KB: [0,8192) L2, [8192,40960) L3
  __shared__ float s_pool[256];
  __shared__ float s_c1w[128];
  __shared__ float s_c1b[64];
  __shared__ float s_c2b[128];
  __shared__ float s_c3b[256];

  const int tid = threadIdx.x;
  // ---- async Global -> LDS staging of the swizzled weight image (80 KB) ----
  {
    uint32_t ldsBase = (uint32_t)(uintptr_t)&s_wh[0];   // addr[31:0] = LDS offset
    const char* gsrc = (const char*)swz;
    for (int i = tid; i < 5120; i += 256) {             // 5120 x 16 B
      uint32_t ldsa = ldsBase + (uint32_t)(i * 16);
      uint64_t ga   = (uint64_t)(uintptr_t)(gsrc + i * 16);
      asm volatile("global_load_async_to_lds_b128 %0, %1, off"
                   :: "v"(ldsa), "v"(ga) : "memory");
    }
    asm volatile("s_wait_asynccnt 0x0" ::: "memory");
  }
  s_pool[tid] = 0.0f;
  if (tid < 128) s_c1w[tid] = c1w[tid];
  if (tid < 64)  s_c1b[tid] = c1b[tid];
  if (tid < 128) s_c2b[tid] = c2b[tid];
  s_c3b[tid] = c3b[tid];
  __syncthreads();

  const int wave = tid >> 5;
  const int lane = tid & 31;
  const int col  = lane & 15;     // point within tile == C/D column
  const int hi   = lane >> 4;     // half-wave
  const float* lx0 = x + 3;
  const float* lx1 = x + 3 + N_POINTS;
  const _Float16* w2l = s_wh + (lane << 4);
  const _Float16* w3l = s_wh + 8192 + (lane << 4);

#pragma unroll 1
  for (int it = 0; it < TILES_PER_WAVE; ++it) {
    const int pbase = ((blockIdx.x * WAVES_PER_BLOCK + wave) * TILES_PER_WAVE + it) * 16;
    const float f0 = lx0[pbase + col];
    const float f1 = lx1[pbase + col];

    // ---- layer 1 (K=2, VALU) straight into WMMA B-fragment layout ----
    F16x16 b1[2];
#pragma unroll
    for (int kc = 0; kc < 2; ++kc) {
      const int cbase = kc * 32 + hi * 16;
#pragma unroll
      for (int j = 0; j < 16; ++j) {
        const int c = cbase + j;
        float v = fmaf(s_c1w[2 * c], f0, fmaf(s_c1w[2 * c + 1], f1, s_c1b[c]));
        b1[kc].h[j] = (_Float16)fmaxf(v, 0.0f);
      }
    }

    // ---- layer 2: 8 tiles of 16 out-channels, K=64 -> 2 WMMA each ----
    uint32_t a16[8][4];   // h2 as packed f16 pairs, C-layout per tile
#pragma unroll
    for (int t2 = 0; t2 < 8; ++t2) {
      v8f acc = {};
#pragma unroll
      for (int kc = 0; kc < 2; ++kc) {
        F16x16 a;
        a.v = *(const v16h*)(w2l + ((t2 * 2 + kc) << 9));
        acc = __builtin_amdgcn_wmma_f32_16x16x32_f16(
                  false, a.v, false, b1[kc].v, (short)0, acc, false, false);
      }
      const int mb = t2 * 16 + hi * 8;
#pragma unroll
      for (int v2 = 0; v2 < 4; ++v2) {
        H2U pk;
        pk.h[0] = (_Float16)fmaxf(acc[2 * v2]     + s_c2b[mb + 2 * v2],     0.0f);
        pk.h[1] = (_Float16)fmaxf(acc[2 * v2 + 1] + s_c2b[mb + 2 * v2 + 1], 0.0f);
        a16[t2][v2] = pk.u;
      }
    }

    // ---- reshape h2 (C-layout) into layer-3 B fragments via half swaps ----
    F16x16 b3[4];
#pragma unroll
    for (int kc = 0; kc < 4; ++kc) {
      const int tLo = 2 * kc, tHi = 2 * kc + 1;
#pragma unroll
      for (int i = 0; i < 4; ++i) {
        uint32_t shLo = (uint32_t)__shfl_xor((int)a16[tLo][i], 16, 32);
        uint32_t shHi = (uint32_t)__shfl_xor((int)a16[tHi][i], 16, 32);
        b3[kc].u[i]     = hi ? shHi        : a16[tLo][i];
        b3[kc].u[i + 4] = hi ? a16[tHi][i] : shLo;
      }
    }

    // ---- layer 3: 16 tiles of 16 out-channels, K=128 -> 4 WMMA each ----
#pragma unroll 2
    for (int t3 = 0; t3 < 16; ++t3) {
      v8f acc = {};
#pragma unroll
      for (int kc = 0; kc < 4; ++kc) {
        F16x16 a;
        a.v = *(const v16h*)(w3l + ((t3 * 4 + kc) << 9));
        acc = __builtin_amdgcn_wmma_f32_16x16x32_f16(
                  false, a.v, false, b3[kc].v, (short)0, acc, false, false);
      }
      const int mb = t3 * 16 + hi * 8;
      float m[8];
#pragma unroll
      for (int v3 = 0; v3 < 8; ++v3)
        m[v3] = fmaxf(acc[v3] + s_c3b[mb + v3], 0.0f);
      // max over the 16 points of this tile (lanes sharing the channel set)
#pragma unroll
      for (int d = 1; d < 16; d <<= 1)
#pragma unroll
        for (int v3 = 0; v3 < 8; ++v3)
          m[v3] = fmaxf(m[v3], __shfl_xor(m[v3], d, 32));
      if (col == 0) {
#pragma unroll
        for (int v3 = 0; v3 < 8; ++v3)
          atomicMax((int*)&s_pool[mb + v3], __float_as_int(m[v3]));
      }
    }
  }
  __syncthreads();
  atomicMax((int*)&gpool[tid], __float_as_int(s_pool[tid]));
}

// ---------------------------------------------------------------------------
// Kernel 2: obs = fcw@pool + fcb, pose MLP, net_input, z = relu(pw@ni + pb)
// ---------------------------------------------------------------------------
__global__ void __launch_bounds__(256) head1_kernel(
    const float* __restrict__ x,
    const float* __restrict__ fcw, const float* __restrict__ fcb,
    const float* __restrict__ ow1, const float* __restrict__ ob1,
    const float* __restrict__ ow2, const float* __restrict__ ob2,
    const float* __restrict__ pw,  const float* __restrict__ pb,
    const float* __restrict__ gpool,
    float* __restrict__ zv, float* __restrict__ out)
{
  __shared__ float s_p[256];
  __shared__ float s_ni[288];
  __shared__ float s_pose1[16];
  const int t = threadIdx.x;
  s_p[t] = gpool[t];
  if (t < 16) {
    float a = ob1[t];
    for (int k = 0; k < 3; ++k) a = fmaf(ow1[t * 3 + k], x[k], a);
    s_pose1[t] = 1.0f / (1.0f + expf(-a));
  }
  __syncthreads();
  {
    float a = fcb[t];
    for (int k = 0; k < 256; ++k) a = fmaf(fcw[t * 256 + k], s_p[k], a);
    s_ni[t] = a;                        // obs has no relu
  }
  if (t < 32) {
    float a = ob2[t];
    for (int k = 0; k < 16; ++k) a = fmaf(ow2[t * 16 + k], s_pose1[k], a);
    s_ni[256 + t] = fmaxf(a, 0.0f);
  }
  __syncthreads();
  out[1031 + t] = s_ni[t];
  if (t < 32) out[1031 + 256 + t] = s_ni[256 + t];
  for (int j = t; j < 512; j += 256) {
    float a = pb[j];
    for (int k = 0; k < 288; ++k) a = fmaf(pw[j * 288 + k], s_ni[k], a);
    zv[j] = fmaxf(a, 0.0f);
  }
}

// ---------------------------------------------------------------------------
// Kernel 3: LSTM gate GEMV (2048 x 512 twice) — one wave per gate row.
// ---------------------------------------------------------------------------
__global__ void __launch_bounds__(256) gates_kernel(
    const float* __restrict__ wih, const float* __restrict__ whh,
    const float* __restrict__ bih, const float* __restrict__ bhh,
    const float* __restrict__ hx,  const float* __restrict__ zv,
    float* __restrict__ gates)
{
  const int wave = threadIdx.x >> 5, lane = threadIdx.x & 31;
  const int j = blockIdx.x * 8 + wave;       // 256 blocks * 8 waves = 2048
  float acc = 0.0f;
  for (int k = lane; k < 512; k += 32)
    acc += wih[j * 512 + k] * zv[k] + whh[j * 512 + k] * hx[k];
  for (int d = 16; d; d >>= 1) acc += __shfl_xor(acc, d, 32);
  if (lane == 0) gates[j] = acc + bih[j] + bhh[j];
}

// ---------------------------------------------------------------------------
// Kernel 4: LSTM cell + q1/q2 + actor/critic heads.
// ---------------------------------------------------------------------------
__global__ void __launch_bounds__(256) head2_kernel(
    const float* __restrict__ cx, const float* __restrict__ gates,
    const float* __restrict__ q1w, const float* __restrict__ q1b,
    const float* __restrict__ q2w, const float* __restrict__ q2b,
    const float* __restrict__ aw1, const float* __restrict__ ab1,
    const float* __restrict__ aw2, const float* __restrict__ ab2,
    const float* __restrict__ iw1, const float* __restrict__ ib1,
    const float* __restrict__ iw2, const float* __restrict__ ib2,
    const float* __restrict__ ew1, const float* __restrict__ eb1,
    const float* __restrict__ ew2, const float* __restrict__ eb2,
    float* __restrict__ out)
{
  __shared__ float s_h[512];
  __shared__ float s_q1[128];
  __shared__ float s_q2[64];
  __shared__ float s_head[48];
  const int t = threadIdx.x;
  for (int j = t; j < 512; j += 256) {
    float ig = 1.0f / (1.0f + expf(-gates[j]));
    float fg = 1.0f / (1.0f + expf(-gates[512 + j]));
    float gg = tanhf(gates[1024 + j]);
    float og = 1.0f / (1.0f + expf(-gates[1536 + j]));
    float cn = fmaf(ig, gg, fg * cx[j]);
    float hn = og * tanhf(cn);
    s_h[j] = hn;
    out[7 + j]   = hn;
    out[519 + j] = cn;
  }
  __syncthreads();
  if (t < 128) {
    float a = q1b[t];
    for (int k = 0; k < 512; ++k) a = fmaf(q1w[t * 512 + k], s_h[k], a);
    s_q1[t] = fmaxf(a, 0.0f);
  }
  __syncthreads();
  if (t < 64) {
    float a = q2b[t];
    for (int k = 0; k < 128; ++k) a = fmaf(q2w[t * 128 + k], s_q1[k], a);
    s_q2[t] = fmaxf(a, 0.0f);
  }
  __syncthreads();
  if (t < 16) {
    float a = ab1[t], ii = ib1[t], e = eb1[t];
    for (int k = 0; k < 64; ++k) {
      a  = fmaf(aw1[t * 64 + k], s_q2[k], a);
      ii = fmaf(iw1[t * 64 + k], s_q2[k], ii);
      e  = fmaf(ew1[t * 64 + k], s_q2[k], e);
    }
    s_head[t]      = fmaxf(a, 0.0f);
    s_head[16 + t] = fmaxf(ii, 0.0f);
    s_head[32 + t] = fmaxf(e, 0.0f);
  }
  __syncthreads();
  if (t == 0) {
    float lg[5], mx = -1e30f;
    for (int j = 0; j < 5; ++j) {
      float a = ab2[j];
      for (int k = 0; k < 16; ++k) a = fmaf(aw2[j * 16 + k], s_head[k], a);
      lg[j] = a; mx = fmaxf(mx, a);
    }
    float sum = 0.0f;
    for (int j = 0; j < 5; ++j) { lg[j] = expf(lg[j] - mx); sum += lg[j]; }
    for (int j = 0; j < 5; ++j) out[j] = lg[j] / sum;
    float ci = ib2[0], ce = eb2[0];
    for (int k = 0; k < 16; ++k) {
      ci = fmaf(iw2[k], s_head[16 + k], ci);
      ce = fmaf(ew2[k], s_head[32 + k], ce);
    }
    out[5] = ci;
    out[6] = ce;
  }
}

// ---------------------------------------------------------------------------
extern "C" void kernel_launch(void* const* d_in, const int* in_sizes, int n_in,
                              void* d_out, int out_size, void* d_ws, size_t ws_size,
                              hipStream_t stream)
{
  const float* x   = (const float*)d_in[0];
  const float* hx  = (const float*)d_in[1];
  const float* cx  = (const float*)d_in[2];
  const float* c1w = (const float*)d_in[3];
  const float* c1b = (const float*)d_in[4];
  const float* c2w = (const float*)d_in[5];
  const float* c2b = (const float*)d_in[6];
  const float* c3w = (const float*)d_in[7];
  const float* c3b = (const float*)d_in[8];
  const float* fcw = (const float*)d_in[9];
  const float* fcb = (const float*)d_in[10];
  const float* ow1 = (const float*)d_in[11];
  const float* ob1 = (const float*)d_in[12];
  const float* ow2 = (const float*)d_in[13];
  const float* ob2 = (const float*)d_in[14];
  const float* pw  = (const float*)d_in[15];
  const float* pb  = (const float*)d_in[16];
  const float* wih = (const float*)d_in[17];
  const float* whh = (const float*)d_in[18];
  const float* bih = (const float*)d_in[19];
  const float* bhh = (const float*)d_in[20];
  const float* q1w = (const float*)d_in[21];
  const float* q1b = (const float*)d_in[22];
  const float* q2w = (const float*)d_in[23];
  const float* q2b = (const float*)d_in[24];
  const float* aw1 = (const float*)d_in[25];
  const float* ab1 = (const float*)d_in[26];
  const float* aw2 = (const float*)d_in[27];
  const float* ab2 = (const float*)d_in[28];
  const float* iw1 = (const float*)d_in[29];
  const float* ib1 = (const float*)d_in[30];
  const float* iw2 = (const float*)d_in[31];
  const float* ib2 = (const float*)d_in[32];
  const float* ew1 = (const float*)d_in[33];
  const float* eb1 = (const float*)d_in[34];
  const float* ew2 = (const float*)d_in[35];
  const float* eb2 = (const float*)d_in[36];

  char* ws = (char*)d_ws;
  float*    gpool = (float*)ws;                      //   256 f32  @ 0
  _Float16* sw2   = (_Float16*)(ws + 1024);          //  8192 f16  @ 1024
  _Float16* sw3   = (_Float16*)(ws + 1024 + 16384);  // 32768 f16  @ 17408 (contiguous with sw2)
  float*    zv    = (float*)(ws + 82944);            //   512 f32
  float*    gts   = (float*)(ws + 84992);            //  2048 f32
  float*    out   = (float*)d_out;

  prep_kernel<<<128, 256, 0, stream>>>(c2w, c3w, sw2, sw3, gpool);
  pointnet_kernel<<<MAIN_BLOCKS, 256, 0, stream>>>(x, c1w, c1b, c2b, c3b,
                                                   sw2, gpool);
  head1_kernel<<<1, 256, 0, stream>>>(x, fcw, fcb, ow1, ob1, ow2, ob2,
                                      pw, pb, gpool, zv, out);
  gates_kernel<<<256, 256, 0, stream>>>(wih, whh, bih, bhh, hx, zv, gts);
  head2_kernel<<<1, 256, 0, stream>>>(cx, gts, q1w, q1b, q2w, q2b,
                                      aw1, ab1, aw2, ab2, iw1, ib1, iw2, ib2,
                                      ew1, eb1, ew2, eb2, out);
}
